// PropModule_5703716569732
// MI455X (gfx1250) — matrix operand order repaired and checked by reading, not compiled, not run on hardware
//
#include <hip/hip_runtime.h>
#include <hip/hip_bf16.h>

typedef __bf16 bf16_t;
typedef __attribute__((ext_vector_type(16))) __bf16 v16bf;
typedef __attribute__((ext_vector_type(8)))  __bf16 v8bf;
typedef __attribute__((ext_vector_type(8)))  float   v8f;

// Problem constants (from reference setup_inputs)
constexpr int BB    = 2;
constexpr int NN    = 2048;
constexpr int RR    = 16384;
constexpr int PFD   = 64;
constexpr int STEPS = 3;

static inline int ceil_div(int a, int b) { return (a + b - 1) / b; }

static __device__ __forceinline__ v16bf cat8(v8bf a, v8bf b) {
    return __builtin_shufflevector(a, b,
        0, 1, 2, 3, 4, 5, 6, 7, 8, 9, 10, 11, 12, 13, 14, 15);
}

// ---------------------------------------------------------------------------
// Recover receiver/sender indices from one-hot matrices: idx = sum_n n * M[b,n,r]
// Coalesced: consecutive threads read consecutive r for each n-row.
// ---------------------------------------------------------------------------
__global__ void extract_idx_kernel(const float* __restrict__ Rr,
                                   const float* __restrict__ Rs,
                                   int* __restrict__ ridx, int* __restrict__ sidx) {
    int t = blockIdx.x * blockDim.x + threadIdx.x;
    if (t >= BB * RR) return;
    int b = t / RR, r = t % RR;
    const float* pr = Rr + (size_t)b * NN * RR + r;
    const float* ps = Rs + (size_t)b * NN * RR + r;
    float ar = 0.0f, as = 0.0f;
    for (int n = 0; n < NN; ++n) {
        ar = fmaf(pr[(size_t)n * RR], (float)n, ar);
        as = fmaf(ps[(size_t)n * RR], (float)n, as);
    }
    ridx[t] = (int)(ar + 0.5f);
    sidx[t] = (int)(as + 0.5f);
}

// ---------------------------------------------------------------------------
// Zero-pad + fp32->bf16 convert (row-major): dst[Rd,Cd] <- src[Rs,Cs]
// Used for A-side activations.
// ---------------------------------------------------------------------------
__global__ void pad_convert_kernel(bf16_t* __restrict__ dst, const float* __restrict__ src,
                                   int Rd, int Cd, int Rsrc, int Csrc) {
    int t = blockIdx.x * blockDim.x + threadIdx.x;
    if (t >= Rd * Cd) return;
    int r = t / Cd, c = t % Cd;
    float v = (r < Rsrc && c < Csrc) ? src[(size_t)r * Csrc + c] : 0.0f;
    dst[t] = (bf16_t)v;
}

// ---------------------------------------------------------------------------
// Pack weight W[Ksrc, N] fp32 into WMMA B-fragment order (zero-pad K to KPAD):
//   dst[(((nT*(KPAD/32) + kc)*32) + lane)*16 + e] = W[k, n]
//   with k = kc*32 + (lane>>4)*16 + e ; n = nT*16 + (lane&15)
// GEMM then loads each fragment as two contiguous b128s.
// ---------------------------------------------------------------------------
__global__ void pack_w_kernel(bf16_t* __restrict__ dst, const float* __restrict__ src,
                              int KPAD, int N, int Ksrc) {
    int t = blockIdx.x * blockDim.x + threadIdx.x;
    if (t >= KPAD * N) return;
    int e     = t & 15;
    int lane  = (t >> 4) & 31;
    int chunk = t >> 9;                 // nT*(KPAD/32) + kc
    int kchunks = KPAD >> 5;
    int nT = chunk / kchunks;
    int kc = chunk - nT * kchunks;
    int k = (kc << 5) + ((lane >> 4) << 4) + e;
    int n = (nT << 4) + (lane & 15);
    float v = (k < Ksrc) ? src[(size_t)k * N + n] : 0.0f;
    dst[t] = (bf16_t)v;
}

__global__ void zero_f32_kernel(float* __restrict__ p, int n) {
    int t = blockIdx.x * blockDim.x + threadIdx.x;
    if (t < n) p[t] = 0.0f;
}

// ---------------------------------------------------------------------------
// Build relation-encoder input [B*R, 32]: [state_r(6) | state_s(6) | Ra(1) | pad]
// ---------------------------------------------------------------------------
__global__ void build_relin_kernel(const float* __restrict__ state,
                                   const float* __restrict__ Ra,
                                   const int* __restrict__ ridx,
                                   const int* __restrict__ sidx,
                                   bf16_t* __restrict__ dst) {
    int t = blockIdx.x * blockDim.x + threadIdx.x;
    if (t >= BB * RR * 32) return;
    int rg = t >> 5, c = t & 31;
    int b = rg / RR;
    float v = 0.0f;
    if (c < 6) {
        v = state[((size_t)b * NN + ridx[rg]) * 6 + c];
    } else if (c < 12) {
        v = state[((size_t)b * NN + sidx[rg]) * 6 + (c - 6)];
    } else if (c == 12) {
        v = Ra[rg];  // REL_DIM == 1
    }
    dst[(size_t)rg * 32 + c] = (bf16_t)v;
}

// ---------------------------------------------------------------------------
// Gather effect_r / effect_s (fp32 -> bf16) into rel_in columns 64..191
// ---------------------------------------------------------------------------
__global__ void gather_effects_kernel(const float* __restrict__ effect,
                                      const int* __restrict__ ridx,
                                      const int* __restrict__ sidx,
                                      bf16_t* __restrict__ rel_in) {
    int t = blockIdx.x * blockDim.x + threadIdx.x;
    if (t >= BB * RR * PFD) return;
    int rg = t >> 6, c = t & 63;
    int b = rg / RR;
    const float* eb = effect + (size_t)b * NN * PFD;
    bf16_t* row = rel_in + (size_t)rg * 192;
    row[64 + c]  = (bf16_t)eb[(size_t)ridx[rg] * PFD + c];
    row[128 + c] = (bf16_t)eb[(size_t)sidx[rg] * PFD + c];
}

// ---------------------------------------------------------------------------
// Scatter-add relation_effect into effect_agg (Rr @ relation_effect)
// ---------------------------------------------------------------------------
__global__ void scatter_add_kernel(const bf16_t* __restrict__ rel_eff,
                                   const int* __restrict__ ridx,
                                   float* __restrict__ effect_agg) {
    int t = blockIdx.x * blockDim.x + threadIdx.x;
    if (t >= BB * RR * PFD) return;
    int rg = t >> 6, c = t & 63;
    int b = rg / RR;
    float v = (float)rel_eff[(size_t)rg * PFD + c];
    atomicAdd(effect_agg + ((size_t)b * NN + ridx[rg]) * PFD + c, v);
}

// ---------------------------------------------------------------------------
// Copy effect_agg (fp32) into prop_in columns 64..127 as bf16
// ---------------------------------------------------------------------------
__global__ void agg_to_propin_kernel(const float* __restrict__ agg,
                                     bf16_t* __restrict__ prop_in) {
    int t = blockIdx.x * blockDim.x + threadIdx.x;
    if (t >= BB * NN * PFD) return;
    int m = t >> 6, c = t & 63;
    prop_in[(size_t)m * 128 + 64 + c] = (bf16_t)agg[t];
}

// ---------------------------------------------------------------------------
// WMMA GEMM: out = act(A[M,K](bf16) @ W[K,N](bf16,packed) + bias + resid)
// One wave per 16x32 output strip (two 16x16 tiles, shared A fragment).
// K consumed in chunks of 32 via v_wmma_f32_16x16x32_bf16 (fp32 accumulate).
// Fragment layouts (C/D per CDNA5 ISA; A/B use this code's contiguous k order):
//   A (16x32 bf16): m = lane&15, khalf = (lane>>4)*16;
//                   elem e -> k=khalf+e
//   B packed so each lane's 16 elems are contiguous (two b128 loads)
//   C/D (16x16 f32): vgpr g -> m = (lane>>4)*8 + g ; n = lane&15
// ---------------------------------------------------------------------------
__global__ void gemm_wmma_kernel(const bf16_t* __restrict__ A, int lda,
                                 const bf16_t* __restrict__ Wp,
                                 const float* __restrict__ bias,
                                 const float* __restrict__ resid, int ldr,
                                 bf16_t* __restrict__ outB, int ldoB,
                                 float* __restrict__ outF, int ldoF,
                                 int M, int N, int K, int doRelu) {
    int wave = (blockIdx.x * blockDim.x + threadIdx.x) >> 5;
    int lane = threadIdx.x & 31;
    int nPairs = N >> 5;                      // pairs of 16-wide n-tiles
    int totalUnits = (M >> 4) * nPairs;
    if (wave >= totalUnits) return;           // wave-uniform: EXEC all-ones at WMMA

    int mT = wave / nPairs;
    int nP = wave - mT * nPairs;

    int row   = (mT << 4) + (lane & 15);
    int khalf = (lane >> 4) << 4;             // 0 or 16 (A layout)
    int kch   = K >> 5;

    const bf16_t* arow = A + (size_t)row * lda + khalf;
    const bf16_t* w0p  = Wp + ((size_t)((2 * nP)     * kch) * 32 + lane) * 16;
    const bf16_t* w1p  = Wp + ((size_t)((2 * nP + 1) * kch) * 32 + lane) * 16;

    v8f acc0 = {}, acc1 = {};
    for (int kc = 0; kc < kch; ++kc) {
        const bf16_t* ap = arow + ((size_t)kc << 5);
        v16bf af = cat8(*(const v8bf*)ap, *(const v8bf*)(ap + 8));

        const bf16_t* b0p = w0p + ((size_t)kc << 9);
        v16bf bf0 = cat8(*(const v8bf*)b0p, *(const v8bf*)(b0p + 8));
        const bf16_t* b1p = w1p + ((size_t)kc << 9);
        v16bf bf1 = cat8(*(const v8bf*)b1p, *(const v8bf*)(b1p + 8));

        acc0 = __builtin_amdgcn_wmma_f32_16x16x32_bf16(
            false, af, false, bf0, (short)0, acc0, false, false);
        acc1 = __builtin_amdgcn_wmma_f32_16x16x32_bf16(
            false, af, false, bf1, (short)0, acc1, false, false);
    }

    int mBase = (mT << 4) + ((lane >> 4) << 3);
    #pragma unroll
    for (int j = 0; j < 2; ++j) {
        int ncol = (((2 * nP + j)) << 4) + (lane & 15);
        float bv = bias ? bias[ncol] : 0.0f;
        #pragma unroll
        for (int g = 0; g < 8; ++g) {
            int m = mBase + g;
            float v = (j == 0 ? acc0[g] : acc1[g]) + bv;
            if (resid) v += resid[(size_t)m * ldr + ncol];
            if (doRelu) v = v > 0.0f ? v : 0.0f;
            if (outF) outF[(size_t)m * ldoF + ncol] = v;
            if (outB) outB[(size_t)m * ldoB + ncol] = (bf16_t)v;
        }
    }
}

// ---------------------------------------------------------------------------
// Final tiny layer: pred[m, 0..2] = hidden[m, :64] @ W1[64,3] + b1
// ---------------------------------------------------------------------------
__global__ void final_layer_kernel(const bf16_t* __restrict__ hidden,
                                   const float* __restrict__ W1,
                                   const float* __restrict__ b1,
                                   float* __restrict__ pred) {
    int t = blockIdx.x * blockDim.x + threadIdx.x;
    if (t >= BB * NN * 3) return;
    int m = t / 3, o = t - m * 3;
    const bf16_t* h = hidden + (size_t)m * PFD;
    float acc = b1[o];
    for (int c = 0; c < PFD; ++c) acc = fmaf((float)h[c], W1[c * 3 + o], acc);
    pred[t] = acc;
}

// ---------------------------------------------------------------------------
// Host side
// ---------------------------------------------------------------------------
static char* bump(char*& p, size_t bytes) {
    char* r = p;
    p += (bytes + 255) & ~(size_t)255;
    return r;
}

static void launch_gemm(const bf16_t* A, int lda, const bf16_t* Wp,
                        const float* bias, const float* resid, int ldr,
                        bf16_t* outB, int ldoB, float* outF, int ldoF,
                        int M, int N, int K, int relu, hipStream_t s) {
    int waves = (M / 16) * (N / 32);
    int grid = ceil_div(waves * 32, 256);
    gemm_wmma_kernel<<<grid, 256, 0, s>>>(A, lda, Wp, bias, resid, ldr,
                                          outB, ldoB, outF, ldoF, M, N, K, relu);
}

static void launch_pad(bf16_t* dst, const float* src, int Rd, int Cd, int Rs, int Cs,
                       hipStream_t s) {
    pad_convert_kernel<<<ceil_div(Rd * Cd, 256), 256, 0, s>>>(dst, src, Rd, Cd, Rs, Cs);
}

static void launch_pack_w(bf16_t* dst, const float* src, int KPAD, int N, int Ksrc,
                          hipStream_t s) {
    pack_w_kernel<<<ceil_div(KPAD * N, 256), 256, 0, s>>>(dst, src, KPAD, N, Ksrc);
}

extern "C" void kernel_launch(void* const* d_in, const int* in_sizes, int n_in,
                              void* d_out, int out_size, void* d_ws, size_t ws_size,
                              hipStream_t stream) {
    (void)in_sizes; (void)n_in; (void)out_size; (void)ws_size;

    // Inputs in setup_inputs() dict order; params dict in insertion order (W, b pairs)
    const float* state = (const float*)d_in[0];
    const float* Rr    = (const float*)d_in[1];
    const float* Rs    = (const float*)d_in[2];
    const float* Ra    = (const float*)d_in[3];
    // d_in[4] = steps (fixed at 3 in setup; cannot read device scalar in capture)
    const float* pe0W = (const float*)d_in[5];  const float* pe0b = (const float*)d_in[6];
    const float* pe1W = (const float*)d_in[7];  const float* pe1b = (const float*)d_in[8];
    const float* pe2W = (const float*)d_in[9];  const float* pe2b = (const float*)d_in[10];
    const float* re0W = (const float*)d_in[11]; const float* re0b = (const float*)d_in[12];
    const float* re1W = (const float*)d_in[13]; const float* re1b = (const float*)d_in[14];
    const float* re2W = (const float*)d_in[15]; const float* re2b = (const float*)d_in[16];
    const float* rpW  = (const float*)d_in[17]; const float* rpb  = (const float*)d_in[18];
    const float* ppW  = (const float*)d_in[19]; const float* ppb  = (const float*)d_in[20];
    const float* pr0W = (const float*)d_in[21]; const float* pr0b = (const float*)d_in[22];
    const float* pr1W = (const float*)d_in[23]; const float* pr1b = (const float*)d_in[24];
    float* pred = (float*)d_out;

    const int M_p = BB * NN;   // 4096 particles
    const int M_r = BB * RR;   // 32768 relations

    // Workspace carve-up
    char* p = (char*)d_ws;
    int*    ridx      = (int*)bump(p, (size_t)M_r * 4);
    int*    sidx      = (int*)bump(p, (size_t)M_r * 4);
    bf16_t* stateB    = (bf16_t*)bump(p, (size_t)M_p * 32 * 2);
    bf16_t* w_pe0     = (bf16_t*)bump(p, (size_t)32 * 128 * 2);
    bf16_t* w_pe1     = (bf16_t*)bump(p, (size_t)128 * 128 * 2);
    bf16_t* w_pe2     = (bf16_t*)bump(p, (size_t)128 * 64 * 2);
    bf16_t* w_re0     = (bf16_t*)bump(p, (size_t)32 * 64 * 2);
    bf16_t* w_re1     = (bf16_t*)bump(p, (size_t)64 * 64 * 2);
    bf16_t* w_re2     = (bf16_t*)bump(p, (size_t)64 * 64 * 2);
    bf16_t* w_rp      = (bf16_t*)bump(p, (size_t)192 * 64 * 2);
    bf16_t* w_pp      = (bf16_t*)bump(p, (size_t)128 * 64 * 2);
    bf16_t* w_pr0     = (bf16_t*)bump(p, (size_t)64 * 64 * 2);
    bf16_t* pe_h1     = (bf16_t*)bump(p, (size_t)M_p * 128 * 2);
    bf16_t* pe_h2     = (bf16_t*)bump(p, (size_t)M_p * 128 * 2);
    bf16_t* prop_in   = (bf16_t*)bump(p, (size_t)M_p * 128 * 2);
    bf16_t* relin_enc = (bf16_t*)bump(p, (size_t)M_r * 32 * 2);
    bf16_t* re_h1     = (bf16_t*)bump(p, (size_t)M_r * 64 * 2);
    bf16_t* re_h2     = (bf16_t*)bump(p, (size_t)M_r * 64 * 2);
    bf16_t* rel_in    = (bf16_t*)bump(p, (size_t)M_r * 192 * 2);
    float*  effect    = (float*)bump(p, (size_t)M_p * 64 * 4);
    float*  eff_agg   = (float*)bump(p, (size_t)M_p * 64 * 4);
    // aliases (dead buffers reused)
    bf16_t* rel_eff   = re_h1;   // [M_r, 64], re_h1 dead after re2 GEMM
    bf16_t* effectB   = pe_h1;   // [M_p, 64], pe buffers dead after pe2 GEMM
    bf16_t* hidden    = pe_h2;   // [M_p, 64]

    // 1) Recover gather/scatter indices from one-hot matrices (dominant HBM pass)
    extract_idx_kernel<<<ceil_div(BB * RR, 256), 256, 0, stream>>>(Rr, Rs, ridx, sidx);

    // 2) Convert state (row-major bf16, K pad 6->32) and pack weights into
    //    WMMA B-fragment order (K padded to multiples of 32)
    launch_pad(stateB, state, M_p, 32, M_p, 6, stream);
    launch_pack_w(w_pe0, pe0W, 32, 128, 6, stream);
    launch_pack_w(w_pe1, pe1W, 128, 128, 128, stream);
    launch_pack_w(w_pe2, pe2W, 128, 64, 128, stream);
    launch_pack_w(w_re0, re0W, 32, 64, 13, stream);
    launch_pack_w(w_re1, re1W, 64, 64, 64, stream);
    launch_pack_w(w_re2, re2W, 64, 64, 64, stream);
    launch_pack_w(w_rp,  rpW, 192, 64, 192, stream);
    launch_pack_w(w_pp,  ppW, 128, 64, 128, stream);
    launch_pack_w(w_pr0, pr0W, 64, 64, 64, stream);

    // 3) Relation encoder input [state_r | state_s | Ra | 0-pad]
    build_relin_kernel<<<ceil_div(BB * RR * 32, 256), 256, 0, stream>>>(
        state, Ra, ridx, sidx, relin_enc);

    // 4) Particle encoder: 6->128->128->64 ; final layer writes prop_in cols 0..63
    launch_gemm(stateB, 32, w_pe0, pe0b, nullptr, 0, pe_h1, 128, nullptr, 0,
                M_p, 128, 32, 1, stream);
    launch_gemm(pe_h1, 128, w_pe1, pe1b, nullptr, 0, pe_h2, 128, nullptr, 0,
                M_p, 128, 128, 1, stream);
    launch_gemm(pe_h2, 128, w_pe2, pe2b, nullptr, 0, prop_in, 128, nullptr, 0,
                M_p, 64, 128, 1, stream);

    // 5) Relation encoder: 13->64->64->64 ; final layer writes rel_in cols 0..63
    launch_gemm(relin_enc, 32, w_re0, re0b, nullptr, 0, re_h1, 64, nullptr, 0,
                M_r, 64, 32, 1, stream);
    launch_gemm(re_h1, 64, w_re1, re1b, nullptr, 0, re_h2, 64, nullptr, 0,
                M_r, 64, 64, 1, stream);
    launch_gemm(re_h2, 64, w_re2, re2b, nullptr, 0, rel_in, 192, nullptr, 0,
                M_r, 64, 64, 1, stream);

    // 6) effect = 0
    zero_f32_kernel<<<ceil_div(M_p * 64, 256), 256, 0, stream>>>(effect, M_p * 64);

    // 7) Propagation steps
    for (int s = 0; s < STEPS; ++s) {
        gather_effects_kernel<<<ceil_div(BB * RR * PFD, 256), 256, 0, stream>>>(
            effect, ridx, sidx, rel_in);
        launch_gemm(rel_in, 192, w_rp, rpb, nullptr, 0, rel_eff, 64, nullptr, 0,
                    M_r, 64, 192, 1, stream);
        zero_f32_kernel<<<ceil_div(M_p * 64, 256), 256, 0, stream>>>(eff_agg, M_p * 64);
        scatter_add_kernel<<<ceil_div(BB * RR * PFD, 256), 256, 0, stream>>>(
            rel_eff, ridx, eff_agg);
        agg_to_propin_kernel<<<ceil_div(BB * NN * PFD, 256), 256, 0, stream>>>(
            eff_agg, prop_in);
        // effect = relu(prop_in @ Wpp + bpp + effect)  (residual fused, in-place fp32)
        launch_gemm(prop_in, 128, w_pp, ppb, effect, 64, nullptr, 0, effect, 64,
                    M_p, 64, 128, 1, stream);
    }

    // 8) Predictor: relu(effect @ W0 + b0) @ W1 + b1
    launch_pad(effectB, effect, M_p, 64, M_p, 64, stream);
    launch_gemm(effectB, 64, w_pr0, pr0b, nullptr, 0, hidden, 64, nullptr, 0,
                M_p, 64, 64, 1, stream);
    final_layer_kernel<<<ceil_div(BB * NN * 3, 256), 256, 0, stream>>>(
        hidden, pr1W, pr1b, pred);
}
